// RWKVBlock_37864431682660
// MI455X (gfx1250) — matrix operand order repaired and checked by reading, not compiled, not run on hardware
//
#include <hip/hip_runtime.h>

// RWKV-style block for MI455X (gfx1250, wave32, WMMA + TDM).
// All GEMMs run through v_wmma_f32_16x16x32_f16 (f16 inputs, f32 accum).
// GEMM A-panels are staged to LDS with the Tensor Data Mover (double-buffered,
// s_wait_tensorcnt), B-panels stream from global (L2-resident weights).
// B=2 T=1024 C=1024 H=16 HS=64 ROT=32 TT=1024 FFN=4096.

#define B_   2
#define T_   1024
#define C_   1024
#define H_   16
#define HS_  64
#define ROT_ 32
#define TT_  1024
#define FFN_ 4096
#define M_   (B_ * T_)   // 2048 GEMM rows
#define KCH_ 128         // TDM k-chunk (elements)

typedef __attribute__((ext_vector_type(16))) _Float16 v16h;
typedef __attribute__((ext_vector_type(8)))  _Float16 v8h;
typedef __attribute__((ext_vector_type(8)))  float    v8f;

#if defined(__has_builtin)
#if __has_builtin(__builtin_amdgcn_tensor_load_to_lds) && \
    __has_builtin(__builtin_amdgcn_s_wait_tensorcnt)
#define HAVE_TDM 1
#endif
#endif

// ---------------------------------------------------------------------------
// WMMA fragment loaders (CDNA5 ISA 7.12.2 layouts, wave32).
// A: 16x32 f16, row-major source with contiguous K (stride lda).
//   lane L (l16=L&15, hi=L>>4): M = m0+l16, elements 0..7 = K kb..kb+7,
//   elements 8..15 = K kb+16..kb+23, kb = k0 + hi*8.
// B: 32x16 f16 from a row-major [N][K] array (pre-transposed weight):
//   lane L: N = n0+l16, elements 0..7 = K kb..kb+7,
//   elements 8..15 = K kb+16..kb+23, kb = k0 + hi*8.
// ---------------------------------------------------------------------------
__device__ __forceinline__ v16h load_frag_a(const _Float16* __restrict__ base,
                                            int lda, int m0, int k0) {
  const int lane = threadIdx.x & 31;
  const int l16 = lane & 15, hi = lane >> 4;
  const _Float16* p = base + (size_t)(m0 + l16) * lda + k0 + hi * 8;
  v8h lo = *(const v8h*)p;
  v8h hp = *(const v8h*)(p + 16);
  v16h a;
#pragma unroll
  for (int i = 0; i < 8; ++i) { a[i] = lo[i]; a[i + 8] = hp[i]; }
  return a;
}

__device__ __forceinline__ v16h load_frag_b(const _Float16* __restrict__ base,
                                            int ldb, int n0, int k0) {
  const int lane = threadIdx.x & 31;
  const int l16 = lane & 15, hi = lane >> 4;
  const _Float16* p = base + (size_t)(n0 + l16) * ldb + k0 + hi * 8;
  v8h lo = *(const v8h*)p;
  v8h hp = *(const v8h*)(p + 16);
  v16h b;
#pragma unroll
  for (int i = 0; i < 8; ++i) { b[i] = lo[i]; b[i + 8] = hp[i]; }
  return b;
}

#ifdef HAVE_TDM
typedef unsigned int u32x4_t __attribute__((ext_vector_type(4)));
typedef int          i32x4_t __attribute__((ext_vector_type(4)));
typedef int          i32x8_t __attribute__((ext_vector_type(8)));

// Issue one 2D TDM load: rows x cols f16 tile, global row stride ld (elems),
// into LDS at byte offset lds_off. Descriptor per CDNA5 ISA 8.3/8.4:
// group0: count=1 | lds_addr | global_addr[56:0] | type=2
// group1: data_size=1(2B) | tensor_dim0=ld | tensor_dim1=rows |
//         tile_dim0=cols | tile_dim1=rows | tensor_dim0_stride=ld
__device__ __forceinline__ void tdm_load_2d(const _Float16* g, int ld,
                                            unsigned int lds_off, int rows, int cols) {
  const unsigned long long ga = (unsigned long long)(uintptr_t)g;
  u32x4_t g0;
  g0[0] = (unsigned)__builtin_amdgcn_readfirstlane(1);
  g0[1] = (unsigned)__builtin_amdgcn_readfirstlane((int)lds_off);
  g0[2] = (unsigned)__builtin_amdgcn_readfirstlane((int)(unsigned)(ga & 0xffffffffull));
  g0[3] = (unsigned)__builtin_amdgcn_readfirstlane(
              (int)(((unsigned)(ga >> 32) & 0x01ffffffu) | 0x80000000u));
  i32x8_t g1;
  g1[0] = __builtin_amdgcn_readfirstlane(1 << 16);                     // data_size=2B
  g1[1] = __builtin_amdgcn_readfirstlane((ld & 0xffff) << 16);         // tdim0 lo16
  g1[2] = __builtin_amdgcn_readfirstlane(((ld >> 16) & 0xffff) |
                                         ((rows & 0xffff) << 16));     // tdim0 hi | tdim1 lo
  g1[3] = __builtin_amdgcn_readfirstlane((cols & 0xffff) << 16);       // tile_dim0
  g1[4] = __builtin_amdgcn_readfirstlane(rows & 0xffff);               // tile_dim1
  g1[5] = __builtin_amdgcn_readfirstlane(ld);                          // dim0_stride lo32
  g1[6] = 0;
  g1[7] = 0;
  i32x4_t z4 = {0, 0, 0, 0};
#if __clang_major__ >= 23
  i32x8_t z8 = {0, 0, 0, 0, 0, 0, 0, 0};
  __builtin_amdgcn_tensor_load_to_lds(g0, g1, z4, z4, z8, 0);
#else
  __builtin_amdgcn_tensor_load_to_lds(g0, g1, z4, z4, 0);
#endif
}
#endif  // HAVE_TDM

// ---------------------------------------------------------------------------
// Generic WMMA GEMM: out[M][N] = A[M][K] (f16) @ BT[N][K]^T (f16) + bias.
// Block = 8 waves, tile 32(M) x 512(N); each wave owns 32x64 (8 accumulators,
// 8 WMMAs per 32-wide k-step). A-panel (32 x KCH_) staged to LDS by TDM,
// double-buffered; B streams from global.
// mode 0: out32 = acc+bias
// mode 2: out32 = (acc+bias)*gamma[m%T] + res[m][n]      (Wo epilogue)
// mode 3: out32 = acc+bias + res[m][n]                   (final residual)
// ---------------------------------------------------------------------------
__global__ void wmma_gemm(const _Float16* __restrict__ A,
                          const _Float16* __restrict__ BT,
                          const float* __restrict__ bias,
                          float* __restrict__ out32,
                          _Float16* __restrict__ out16,
                          const float* __restrict__ gamma,
                          const float* __restrict__ res,
                          int M, int N, int K, int mode) {
  const int wave = threadIdx.x >> 5;
  const int lane = threadIdx.x & 31;
  const int m0 = blockIdx.x * 32;
  const int n0 = blockIdx.y * 512 + wave * 64;

  v8f acc[2][4] = {};

#ifdef HAVE_TDM
  __shared__ _Float16 Ash[2][32 * KCH_];  // 2 x 8KB double buffer
  const unsigned int lds_off[2] = {(unsigned int)(uintptr_t)&Ash[0][0],
                                   (unsigned int)(uintptr_t)&Ash[1][0]};
  const int nchunk = K / KCH_;
  if (threadIdx.x < 32)  // wave 0 drives the TDM
    tdm_load_2d(A + (size_t)m0 * K, K, lds_off[0], 32, KCH_);
  for (int c = 0; c < nchunk; ++c) {
    if (threadIdx.x < 32) {
      if (c + 1 < nchunk) {
        tdm_load_2d(A + (size_t)m0 * K + (c + 1) * KCH_, K,
                    lds_off[(c + 1) & 1], 32, KCH_);
        __builtin_amdgcn_s_wait_tensorcnt(1);   // chunk c landed
      } else {
        __builtin_amdgcn_s_wait_tensorcnt(0);
      }
    }
    __syncthreads();                            // chunk c visible to all waves
    const _Float16* Ab = Ash[c & 1];
#pragma unroll
    for (int ks = 0; ks < KCH_; ks += 32) {
      v16h a0 = load_frag_a(Ab, KCH_, 0, ks);   // ds_load from LDS
      v16h a1 = load_frag_a(Ab, KCH_, 16, ks);
      const int k0 = c * KCH_ + ks;
#pragma unroll
      for (int t = 0; t < 4; ++t) {
        v16h b = load_frag_b(BT, K, n0 + 16 * t, k0);
        acc[0][t] = __builtin_amdgcn_wmma_f32_16x16x32_f16(false, a0, false, b, (short)0, acc[0][t], false, false);
        acc[1][t] = __builtin_amdgcn_wmma_f32_16x16x32_f16(false, a1, false, b, (short)0, acc[1][t], false, false);
      }
    }
    __syncthreads();                            // buffer free for chunk c+2
  }
#else
  for (int k0 = 0; k0 < K; k0 += 32) {
    v16h a0 = load_frag_a(A, K, m0, k0);
    v16h a1 = load_frag_a(A, K, m0 + 16, k0);
#pragma unroll
    for (int t = 0; t < 4; ++t) {
      v16h b = load_frag_b(BT, K, n0 + 16 * t, k0);
      acc[0][t] = __builtin_amdgcn_wmma_f32_16x16x32_f16(false, a0, false, b, (short)0, acc[0][t], false, false);
      acc[1][t] = __builtin_amdgcn_wmma_f32_16x16x32_f16(false, a1, false, b, (short)0, acc[1][t], false, false);
    }
  }
#endif

  const int l16 = lane & 15, hi = lane >> 4;
#pragma unroll
  for (int mi = 0; mi < 2; ++mi) {
#pragma unroll
    for (int t = 0; t < 4; ++t) {
      const int n = n0 + 16 * t + l16;
#pragma unroll
      for (int r = 0; r < 8; ++r) {
        const int m = m0 + mi * 16 + r + 8 * hi;
        float v = acc[mi][t][r] + (bias ? bias[n] : 0.f);
        const size_t o = (size_t)m * N + n;
        if (mode == 2)      v = v * gamma[m % T_] + res[o];
        else if (mode == 3) v = v + res[o];
        if (out32) out32[o] = v;
        if (out16) out16[o] = (_Float16)v;
      }
    }
  }
}

// fp32 [K][N] weight -> f16 transposed [N][K]
__global__ void transpose_h(const float* __restrict__ W, _Float16* __restrict__ WT,
                            int K, int N) {
  const size_t idx = (size_t)blockIdx.x * 256 + threadIdx.x;
  if (idx >= (size_t)K * N) return;
  const int n = (int)(idx / K), k = (int)(idx % K);
  WT[idx] = (_Float16)W[(size_t)k * N + n];
}

// LayerNorm over C per row (block = row). Optional f32 and/or f16 output.
__global__ void layernorm(const float* __restrict__ x, const float* __restrict__ g,
                          const float* __restrict__ b,
                          float* __restrict__ out32, _Float16* __restrict__ out16) {
  const int row = blockIdx.x;  // [0, B*T)
  const float* xr = x + (size_t)row * C_;
  __shared__ float red[256];

  float s = 0.f;
  for (int c = threadIdx.x; c < C_; c += 256) s += xr[c];
  red[threadIdx.x] = s; __syncthreads();
  for (int o = 128; o > 0; o >>= 1) {
    if (threadIdx.x < o) red[threadIdx.x] += red[threadIdx.x + o];
    __syncthreads();
  }
  const float mean = red[0] * (1.0f / C_); __syncthreads();

  float v = 0.f;
  for (int c = threadIdx.x; c < C_; c += 256) {
    const float d = xr[c] - mean; v += d * d;
  }
  red[threadIdx.x] = v; __syncthreads();
  for (int o = 128; o > 0; o >>= 1) {
    if (threadIdx.x < o) red[threadIdx.x] += red[threadIdx.x + o];
    __syncthreads();
  }
  const float rstd = rsqrtf(red[0] * (1.0f / C_) + 1e-6f);

  for (int c = threadIdx.x; c < C_; c += 256) {
    const float y = (xr[c] - mean) * rstd * g[c] + b[c];
    const size_t o = (size_t)row * C_ + c;
    if (out32) out32[o] = y;
    if (out16) out16[o] = (_Float16)y;
  }
}

// time_shift_half: low half channels shifted by one timestep; f16 output.
__global__ void time_shift(const float* __restrict__ ln, _Float16* __restrict__ xs) {
  const size_t idx = (size_t)blockIdx.x * 256 + threadIdx.x;  // B*T*C
  if (idx >= (size_t)B_ * T_ * C_) return;
  const int c = (int)(idx % C_);
  const int t = (int)((idx / C_) % T_);
  float v;
  if (c < C_ / 2) v = (t == 0) ? 0.f : ln[idx - C_];
  else            v = ln[idx];
  xs[idx] = (_Float16)v;
}

// Rotary on first ROT dims of q,k; pack q,k -> [b][h][t][d] f16,
// v -> transposed [b][h][d][t] f16 (so att@v B-fragments read contiguous K=u).
__global__ void rotary_pack(const float* __restrict__ q32, const float* __restrict__ k32,
                            const float* __restrict__ v32,
                            _Float16* __restrict__ qh, _Float16* __restrict__ kh,
                            _Float16* __restrict__ vT) {
  const size_t idx = (size_t)blockIdx.x * 256 + threadIdx.x;  // B*T*C
  if (idx >= (size_t)B_ * T_ * C_) return;
  const int d = (int)(idx % HS_);
  const int h = (int)((idx / HS_) % H_);
  const int t = (int)((idx / C_) % T_);
  const int b = (int)(idx / ((size_t)T_ * C_));

  float qv = q32[idx], kv = k32[idx], vv = v32[idx];
  if (d < ROT_) {
    const int i = d & 15;  // emb = concat(freqs, freqs)
    const float invf = __powf((float)TT_, -((float)(2 * i) / (float)ROT_));
    const float ang = (float)t * invf;
    const float cs = __cosf(ang), sn = __sinf(ang);
    const float qo = (d < 16) ? -q32[idx + 16] : q32[idx - 16];
    const float ko = (d < 16) ? -k32[idx + 16] : k32[idx - 16];
    qv = qv * cs + qo * sn;
    kv = kv * cs + ko * sn;
  }
  const size_t o = (((size_t)b * H_ + h) * T_ + t) * HS_ + d;
  qh[o] = (_Float16)qv;
  kh[o] = (_Float16)kv;
  vT[(((size_t)b * H_ + h) * HS_ + d) * T_ + t] = (_Float16)vv;
}

// Scores: S[b,h,t,u] = (q.k)/sqrt(HS), causal (-inf above diagonal).
// Wave tile 32(t) x 64(u), WMMA over d (K=HS=64).
__global__ void attn_scores(const _Float16* __restrict__ qh,
                            const _Float16* __restrict__ kh,
                            float* __restrict__ S) {
  const int bh = blockIdx.z;
  const int wave = threadIdx.x >> 5, lane = threadIdx.x & 31;
  const int t0 = blockIdx.x * 32;
  const int u0 = blockIdx.y * 512 + wave * 64;
  const _Float16* q = qh + (size_t)bh * T_ * HS_;
  const _Float16* k = kh + (size_t)bh * T_ * HS_;

  v8f acc[2][4] = {};
#pragma unroll
  for (int k0 = 0; k0 < HS_; k0 += 32) {
    v16h a0 = load_frag_a(q, HS_, t0, k0);
    v16h a1 = load_frag_a(q, HS_, t0 + 16, k0);
#pragma unroll
    for (int t = 0; t < 4; ++t) {
      v16h b = load_frag_b(k, HS_, u0 + 16 * t, k0);
      acc[0][t] = __builtin_amdgcn_wmma_f32_16x16x32_f16(false, a0, false, b, (short)0, acc[0][t], false, false);
      acc[1][t] = __builtin_amdgcn_wmma_f32_16x16x32_f16(false, a1, false, b, (short)0, acc[1][t], false, false);
    }
  }

  float* Sp = S + (size_t)bh * T_ * T_;
  const float scale = 0.125f;  // 1/sqrt(64)
  const int l16 = lane & 15, hi = lane >> 4;
#pragma unroll
  for (int mi = 0; mi < 2; ++mi) {
#pragma unroll
    for (int t = 0; t < 4; ++t) {
      const int u = u0 + 16 * t + l16;
#pragma unroll
      for (int r = 0; r < 8; ++r) {
        const int tt = t0 + mi * 16 + r + 8 * hi;
        Sp[(size_t)tt * T_ + u] = (u <= tt) ? acc[mi][t][r] * scale : -__builtin_inff();
      }
    }
  }
}

// Row softmax fused with RWKV decay w[h,t,u] = time_w[h,TT-1+u-t]*alpha[h,u]*beta[h,t].
__global__ void softmax_w(const float* __restrict__ S, _Float16* __restrict__ P,
                          const float* __restrict__ time_w,
                          const float* __restrict__ alpha,
                          const float* __restrict__ beta) {
  const int row = blockIdx.x;  // b*H*T + h*T + t
  const int h = (row / T_) % H_;
  const int t = row % T_;
  const float* s = S + (size_t)row * T_;
  __shared__ float red[256];

  float mx = -__builtin_inff();
  for (int u = threadIdx.x; u < T_; u += 256) mx = fmaxf(mx, s[u]);
  red[threadIdx.x] = mx; __syncthreads();
  for (int o = 128; o > 0; o >>= 1) {
    if (threadIdx.x < o) red[threadIdx.x] = fmaxf(red[threadIdx.x], red[threadIdx.x + o]);
    __syncthreads();
  }
  mx = red[0]; __syncthreads();

  float sum = 0.f;
  for (int u = threadIdx.x; u < T_; u += 256) sum += __expf(s[u] - mx);
  red[threadIdx.x] = sum; __syncthreads();
  for (int o = 128; o > 0; o >>= 1) {
    if (threadIdx.x < o) red[threadIdx.x] += red[threadIdx.x + o];
    __syncthreads();
  }
  const float inv = 1.0f / red[0];
  const float bt = beta[h * TT_ + t];

  _Float16* p = P + (size_t)row * T_;
  for (int u = threadIdx.x; u < T_; u += 256) {
    float w = 0.f;
    if (u <= t) w = time_w[h * TT_ + (TT_ - 1 + u - t)] * alpha[h * TT_ + u] * bt;
    p[u] = (_Float16)(__expf(s[u] - mx) * inv * w);
  }
}

// Head mix: A2[b,i,t,u] = sum_j Wmix[i,j] * P[b,j,t,u]
__global__ void head_mix(const _Float16* __restrict__ P, _Float16* __restrict__ A2,
                         const float* __restrict__ Wmix) {
  __shared__ float wm[H_ * H_];
  wm[threadIdx.x & 255] = Wmix[threadIdx.x & 255];
  __syncthreads();
  const size_t idx = (size_t)blockIdx.x * 256 + threadIdx.x;  // B*T*T
  if (idx >= (size_t)B_ * T_ * T_) return;
  const int b = (int)(idx / ((size_t)T_ * T_));
  const size_t tu = idx % ((size_t)T_ * T_);

  float pj[H_];
  const _Float16* pb = P + (size_t)b * H_ * T_ * T_ + tu;
#pragma unroll
  for (int j = 0; j < H_; ++j) pj[j] = (float)pb[(size_t)j * T_ * T_];
  _Float16* ob = A2 + (size_t)b * H_ * T_ * T_ + tu;
#pragma unroll
  for (int i = 0; i < H_; ++i) {
    float a = 0.f;
#pragma unroll
    for (int j = 0; j < H_; ++j) a += wm[i * H_ + j] * pj[j];
    ob[(size_t)i * T_ * T_] = (_Float16)a;
  }
}

// y[b,t,h*HS+d] = sum_u A2[b,h,t,u] * v[b,h,u,d]. Wave tile 32(t) x 64(d).
__global__ void attn_y(const _Float16* __restrict__ A2, const _Float16* __restrict__ vT,
                       _Float16* __restrict__ y) {
  const int bh = blockIdx.z;
  const int b = bh / H_, h = bh % H_;
  const int wave = threadIdx.x >> 5, lane = threadIdx.x & 31;
  const int t0 = (blockIdx.x * 8 + wave) * 32;
  const _Float16* A = A2 + (size_t)bh * T_ * T_;
  const _Float16* Vt = vT + (size_t)bh * HS_ * T_;

  v8f acc[2][4] = {};
  for (int k0 = 0; k0 < T_; k0 += 32) {
    v16h a0 = load_frag_a(A, T_, t0, k0);
    v16h a1 = load_frag_a(A, T_, t0 + 16, k0);
#pragma unroll
    for (int n = 0; n < 4; ++n) {
      v16h b = load_frag_b(Vt, T_, n * 16, k0);
      acc[0][n] = __builtin_amdgcn_wmma_f32_16x16x32_f16(false, a0, false, b, (short)0, acc[0][n], false, false);
      acc[1][n] = __builtin_amdgcn_wmma_f32_16x16x32_f16(false, a1, false, b, (short)0, acc[1][n], false, false);
    }
  }

  const int l16 = lane & 15, hi = lane >> 4;
#pragma unroll
  for (int mi = 0; mi < 2; ++mi) {
#pragma unroll
    for (int n = 0; n < 4; ++n) {
      const int d = n * 16 + l16;
#pragma unroll
      for (int r = 0; r < 8; ++r) {
        const int t = t0 + mi * 16 + r + 8 * hi;
        y[((size_t)(b * T_ + t)) * C_ + h * HS_ + d] = (_Float16)acc[mi][n][r];
      }
    }
  }
}

// g = gelu_exact(kk) * vv  -> f16
__global__ void gelu_mul(const float* __restrict__ kk, const float* __restrict__ vv,
                         _Float16* __restrict__ g) {
  const size_t idx = (size_t)blockIdx.x * 256 + threadIdx.x;  // B*T*FFN
  if (idx >= (size_t)B_ * T_ * FFN_) return;
  const float x = kk[idx];
  const float ge = 0.5f * x * (1.0f + erff(x * 0.70710678118654752f));
  g[idx] = (_Float16)(ge * vv[idx]);
}

// ---------------------------------------------------------------------------
extern "C" void kernel_launch(void* const* d_in, const int* in_sizes, int n_in,
                              void* d_out, int out_size, void* d_ws, size_t ws_size,
                              hipStream_t stream) {
  (void)in_sizes; (void)n_in; (void)out_size; (void)ws_size;
  const float* x     = (const float*)d_in[0];
  const float* ln1g  = (const float*)d_in[1];
  const float* ln1b  = (const float*)d_in[2];
  const float* Wq    = (const float*)d_in[3];
  const float* bq    = (const float*)d_in[4];
  const float* Wk    = (const float*)d_in[5];
  const float* bk    = (const float*)d_in[6];
  const float* Wv    = (const float*)d_in[7];
  const float* bv    = (const float*)d_in[8];
  const float* Wo    = (const float*)d_in[9];
  const float* bo    = (const float*)d_in[10];
  const float* timew = (const float*)d_in[11];
  const float* alpha = (const float*)d_in[12];
  const float* beta  = (const float*)d_in[13];
  const float* gamma = (const float*)d_in[14];
  const float* Wmix  = (const float*)d_in[15];
  const float* ln2g  = (const float*)d_in[16];
  const float* ln2b  = (const float*)d_in[17];
  const float* Wgk   = (const float*)d_in[18];
  const float* bgk   = (const float*)d_in[19];
  const float* Wgv   = (const float*)d_in[20];
  const float* bgv   = (const float*)d_in[21];
  const float* Wgw   = (const float*)d_in[22];
  const float* bgw   = (const float*)d_in[23];
  float* out = (float*)d_out;

  // Workspace bump allocator (256B aligned regions). Total ~450 MB.
  char* ws = (char*)d_ws;
  size_t off = 0;
  auto alloc = [&](size_t bytes) -> char* {
    char* p = ws + off;
    off = (off + bytes + 255) & ~(size_t)255;
    return p;
  };
  _Float16* WqT  = (_Float16*)alloc((size_t)C_ * C_ * 2);
  _Float16* WkT  = (_Float16*)alloc((size_t)C_ * C_ * 2);
  _Float16* WvT  = (_Float16*)alloc((size_t)C_ * C_ * 2);
  _Float16* WoT  = (_Float16*)alloc((size_t)C_ * C_ * 2);
  _Float16* WgkT = (_Float16*)alloc((size_t)C_ * FFN_ * 2);
  _Float16* WgvT = (_Float16*)alloc((size_t)C_ * FFN_ * 2);
  _Float16* WgwT = (_Float16*)alloc((size_t)FFN_ * C_ * 2);
  float*    ln1o = (float*)   alloc((size_t)M_ * C_ * 4);
  _Float16* xs   = (_Float16*)alloc((size_t)M_ * C_ * 2);
  float*    q32  = (float*)   alloc((size_t)M_ * C_ * 4);
  float*    k32  = (float*)   alloc((size_t)M_ * C_ * 4);
  float*    v32  = (float*)   alloc((size_t)M_ * C_ * 4);
  _Float16* qh   = (_Float16*)alloc((size_t)M_ * C_ * 2);
  _Float16* kh   = (_Float16*)alloc((size_t)M_ * C_ * 2);
  _Float16* vT   = (_Float16*)alloc((size_t)M_ * C_ * 2);
  float*    S    = (float*)   alloc((size_t)B_ * H_ * T_ * T_ * 4);
  _Float16* P    = (_Float16*)alloc((size_t)B_ * H_ * T_ * T_ * 2);
  _Float16* A2   = (_Float16*)alloc((size_t)B_ * H_ * T_ * T_ * 2);
  _Float16* yh   = (_Float16*)alloc((size_t)M_ * C_ * 2);
  float*    x1   = (float*)   alloc((size_t)M_ * C_ * 4);
  _Float16* xm   = (_Float16*)alloc((size_t)M_ * C_ * 2);
  float*    kk   = (float*)   alloc((size_t)M_ * FFN_ * 4);
  float*    vvb  = (float*)   alloc((size_t)M_ * FFN_ * 4);
  _Float16* gh   = (_Float16*)alloc((size_t)M_ * FFN_ * 2);

  const dim3 blk(256);

  // 1) f16 transposed weights
  transpose_h<<<dim3((C_ * C_) / 256), blk, 0, stream>>>(Wq,  WqT,  C_, C_);
  transpose_h<<<dim3((C_ * C_) / 256), blk, 0, stream>>>(Wk,  WkT,  C_, C_);
  transpose_h<<<dim3((C_ * C_) / 256), blk, 0, stream>>>(Wv,  WvT,  C_, C_);
  transpose_h<<<dim3((C_ * C_) / 256), blk, 0, stream>>>(Wo,  WoT,  C_, C_);
  transpose_h<<<dim3((C_ * FFN_) / 256), blk, 0, stream>>>(Wgk, WgkT, C_, FFN_);
  transpose_h<<<dim3((C_ * FFN_) / 256), blk, 0, stream>>>(Wgv, WgvT, C_, FFN_);
  transpose_h<<<dim3((FFN_ * C_) / 256), blk, 0, stream>>>(Wgw, WgwT, FFN_, C_);

  // 2) LN1 + time shift
  layernorm<<<dim3(M_), blk, 0, stream>>>(x, ln1g, ln1b, ln1o, nullptr);
  time_shift<<<dim3((M_ * C_) / 256), blk, 0, stream>>>(ln1o, xs);

  // 3) QKV projections (WMMA + TDM staging)
  wmma_gemm<<<dim3(M_ / 32, C_ / 512), blk, 0, stream>>>(xs, WqT, bq, q32, nullptr, nullptr, nullptr, M_, C_, C_, 0);
  wmma_gemm<<<dim3(M_ / 32, C_ / 512), blk, 0, stream>>>(xs, WkT, bk, k32, nullptr, nullptr, nullptr, M_, C_, C_, 0);
  wmma_gemm<<<dim3(M_ / 32, C_ / 512), blk, 0, stream>>>(xs, WvT, bv, v32, nullptr, nullptr, nullptr, M_, C_, C_, 0);

  // 4) rotary + pack
  rotary_pack<<<dim3((M_ * C_) / 256), blk, 0, stream>>>(q32, k32, v32, qh, kh, vT);

  // 5) scores, 6) softmax*w, 7) head mix, att@v
  attn_scores<<<dim3(T_ / 32, T_ / 512, B_ * H_), blk, 0, stream>>>(qh, kh, S);
  softmax_w<<<dim3(B_ * H_ * T_), blk, 0, stream>>>(S, P, timew, alpha, beta);
  head_mix<<<dim3((B_ * T_ * T_) / 256), blk, 0, stream>>>(P, A2, Wmix);
  attn_y<<<dim3(T_ / 256, 1, B_ * H_), blk, 0, stream>>>(A2, vT, yh);

  // 8) Wo + gamma + residual -> x1
  wmma_gemm<<<dim3(M_ / 32, C_ / 512), blk, 0, stream>>>(yh, WoT, bo, x1, nullptr, gamma, x, M_, C_, C_, 2);

  // 9) LN2 -> xm (f16), FFN
  layernorm<<<dim3(M_), blk, 0, stream>>>(x1, ln2g, ln2b, nullptr, xm);
  wmma_gemm<<<dim3(M_ / 32, FFN_ / 512), blk, 0, stream>>>(xm, WgkT, bgk, kk,  nullptr, nullptr, nullptr, M_, FFN_, C_, 0);
  wmma_gemm<<<dim3(M_ / 32, FFN_ / 512), blk, 0, stream>>>(xm, WgvT, bgv, vvb, nullptr, nullptr, nullptr, M_, FFN_, C_, 0);
  gelu_mul<<<dim3((M_ * FFN_) / 256), blk, 0, stream>>>(kk, vvb, gh);
  wmma_gemm<<<dim3(M_ / 32, C_ / 512), blk, 0, stream>>>(gh, WgwT, bgw, out, nullptr, nullptr, x1, M_, C_, FFN_, 3);
}
